// Corr_pyTorch_19705309954343
// MI455X (gfx1250) — matrix operand order repaired and checked by reading, not compiled, not run on hardware
//
#include <hip/hip_runtime.h>

typedef _Float16 v16h __attribute__((ext_vector_type(16)));
typedef _Float16 v8h  __attribute__((ext_vector_type(8)));
typedef float    v8f  __attribute__((ext_vector_type(8)));

#define BATCH  8
#define CH     192
#define HH     128
#define WW     128
#define MAXD   4
#define WIN    9
#define NDISP  (WIN * WIN)           // 81
#define TILE_X 16
#define SLAB_X 24                    // TILE_X + 2*MAXD
#define KC     96                    // channels per LDS phase (2 phases)
#define CPAD   104                   // padded channel stride in halves (208B, 16B-aligned)
#define NTHREADS 288                 // 9 waves: one wave per dy row

__global__ __launch_bounds__(NTHREADS) void corr_wmma_kernel(
    const float* __restrict__ in1,
    const float* __restrict__ in2,
    float* __restrict__ out)
{
    // LDS: in2 slab [dy 9][x 24][c 96(pad 104)], in1 tile [x 16][c 96(pad 104)]
    __shared__ _Float16 sB[WIN * SLAB_X * CPAD];   // 44928 B
    __shared__ _Float16 sA[TILE_X * CPAD];         //  3328 B

    const int g   = blockIdx.x;        // 0 .. 8*128*8-1
    const int b   = g >> 10;           // / (128*8)
    const int rem = g & 1023;
    const int y   = rem >> 3;
    const int x0  = (rem & 7) << 4;

    const int tid  = threadIdx.x;
    const int lane = tid & 31;
    const int wav  = tid >> 5;         // 0..8  == dy row this wave owns
    const int n    = lane & 15;        // N column within 16x16 tile
    const int hi   = lane >> 4;        // half-wave selector
    const int dy   = wav;

    v8f acc0 = {};   // N-tile 0: slab columns 0..15
    v8f acc1 = {};   // N-tile 1: slab columns 16..23 (cols 24..31 zero)

    for (int phase = 0; phase < 2; ++phase) {
        const int c0 = phase * KC;
        __syncthreads();   // previous phase's compute must finish before overwrite

        // ---- cooperative load: in2 slab chunk (f32 -> f16, zero-padded) ----
        for (int i = tid; i < WIN * KC * SLAB_X; i += NTHREADS) {
            int row = i / (KC * SLAB_X);
            int r2  = i - row * (KC * SLAB_X);
            int c   = r2 / SLAB_X;
            int x   = r2 - c * SLAB_X;
            int y2  = y + row - MAXD;
            int gx  = x0 - MAXD + x;
            float v = 0.0f;
            if ((unsigned)y2 < (unsigned)HH && (unsigned)gx < (unsigned)WW)
                v = in2[(((b * CH + c0 + c) * HH + y2) * WW) + gx];
            sB[(row * SLAB_X + x) * CPAD + c] = (_Float16)v;
        }
        // ---- cooperative load: in1 tile chunk ----
        for (int i = tid; i < KC * TILE_X; i += NTHREADS) {
            int c = i >> 4;
            int x = i & 15;
            sA[x * CPAD + c] =
                (_Float16)in1[(((b * CH + c0 + c) * HH + y) * WW) + x0 + x];
        }
        __syncthreads();

        // ---- WMMA: 3 K-steps of 32 channels each, fully unrolled ----
#pragma unroll
        for (int kc = 0; kc < 3; ++kc) {
            // A fragment: lane<16 holds K{0..15}, lane>=16 holds K{16..31}
            const _Float16* ap = &sA[(lane & 15) * CPAD + kc * 32 + hi * 16];
            v8h alo = *(const v8h*)(ap);
            v8h ahi = *(const v8h*)(ap + 8);
            v16h afr = __builtin_shufflevector(alo, ahi,
                0,1,2,3,4,5,6,7,8,9,10,11,12,13,14,15);

            // B fragment, N-tile 0 (columns 0..15): lane = column
            const _Float16* bp0 = &sB[(dy * SLAB_X + n) * CPAD + kc * 32 + hi * 16];
            v8h b0lo = *(const v8h*)(bp0);
            v8h b0hi = *(const v8h*)(bp0 + 8);
            v16h bfr0 = __builtin_shufflevector(b0lo, b0hi,
                0,1,2,3,4,5,6,7,8,9,10,11,12,13,14,15);
            acc0 = __builtin_amdgcn_wmma_f32_16x16x32_f16(
                false, afr, false, bfr0, (short)0, acc0, false, false);

            // B fragment, N-tile 1 (columns 16..23; 24..31 are zeros)
            v16h bfr1 = {};
            if (n < SLAB_X - 16) {
                const _Float16* bp1 =
                    &sB[(dy * SLAB_X + n + 16) * CPAD + kc * 32 + hi * 16];
                v8h b1lo = *(const v8h*)(bp1);
                v8h b1hi = *(const v8h*)(bp1 + 8);
                bfr1 = __builtin_shufflevector(b1lo, b1hi,
                    0,1,2,3,4,5,6,7,8,9,10,11,12,13,14,15);
            }
            acc1 = __builtin_amdgcn_wmma_f32_16x16x32_f16(
                false, afr, false, bfr1, (short)0, acc1, false, false);
        }
    }

    // ---- epilogue: scatter the valid band P[m, m+dx] ----
    const float scale = 1.0f / (float)CH;
#pragma unroll
    for (int r = 0; r < 8; ++r) {
        const int m = r + (hi << 3);     // output pixel within tile
        {
            int dx = n - m;              // N-tile 0: j = n
            if ((unsigned)dx < (unsigned)WIN) {
                int d = dy * WIN + dx;
                out[(((b * NDISP + d) * HH + y) * WW) + x0 + m] = acc0[r] * scale;
            }
        }
        {
            int dx = n + 16 - m;         // N-tile 1: j = n + 16
            if ((unsigned)dx < (unsigned)WIN) {
                int d = dy * WIN + dx;
                out[(((b * NDISP + d) * HH + y) * WW) + x0 + m] = acc1[r] * scale;
            }
        }
    }
}

extern "C" void kernel_launch(void* const* d_in, const int* in_sizes, int n_in,
                              void* d_out, int out_size, void* d_ws, size_t ws_size,
                              hipStream_t stream) {
    (void)in_sizes; (void)n_in; (void)out_size; (void)d_ws; (void)ws_size;
    const float* in1 = (const float*)d_in[0];
    const float* in2 = (const float*)d_in[1];
    float* out = (float*)d_out;

    dim3 grid(BATCH * HH * (WW / TILE_X));  // 8192 workgroups
    dim3 block(NTHREADS);                    // 9 waves (wave32), one per dy
    corr_wmma_kernel<<<grid, block, 0, stream>>>(in1, in2, out);
}
